// DNAGATv2Block_40398462386462
// MI455X (gfx1250) — matrix operand order repaired and checked by reading, not compiled, not run on hardware
//
#include <hip/hip_runtime.h>
#include <hip/hip_bf16.h>

// ---------------- problem constants ----------------
#define N_NODES 50000
#define F_IN    128
#define HC      128      // H*C
#define NHEAD   4
#define E_EDGES 800000
#define ET      (E_EDGES + N_NODES)   // edges + self loops = 850000
#define NEG_SLOPE 0.2f
#define GN_EPS  1e-5f

typedef __attribute__((ext_vector_type(16))) _Float16 v16h;
typedef __attribute__((ext_vector_type(8)))  _Float16 v8h;
typedef __attribute__((ext_vector_type(8)))  float    v8f;

// ---------------- helpers ----------------
__device__ __forceinline__ void edge_sd(const int* __restrict__ ei, int e, int& s, int& d) {
    if (e < E_EDGES) { s = ei[e]; d = ei[E_EDGES + e]; }
    else             { s = e - E_EDGES; d = s; }   // self loop
}

// order-preserving f32 <-> u32 encoding for atomicMax over signed floats
__device__ __forceinline__ unsigned enc_f32(float x) {
    unsigned u = __float_as_uint(x);
    return (u & 0x80000000u) ? ~u : (u | 0x80000000u);
}
__device__ __forceinline__ float dec_f32(unsigned e) {
    unsigned u = (e & 0x80000000u) ? (e & 0x7FFFFFFFu) : ~e;
    return __uint_as_float(u);
}
#define ENC_NEG_INF 0x007FFFFFu   // enc(-inf)

__device__ __forceinline__ float lrelu(float v) { return v > 0.f ? v : NEG_SLOPE * v; }

// ---------------- kernel 0a: f32 -> f16 convert of x ----------------
__global__ void k_conv_f16(const float* __restrict__ x, _Float16* __restrict__ x16) {
    size_t i = (size_t)blockIdx.x * blockDim.x + threadIdx.x;
    if (i < (size_t)N_NODES * F_IN) x16[i] = (_Float16)x[i];
}

// ---------------- kernel 0b: pre-swizzle W into WMMA B-fragments ----------------
// frag layout: fidx = kc*8 + nt ; per frag 32 lanes x 16 f16 contiguous.
// lane L holds col n0 + L%16 ; element i holds K = k0 + (L/16)*8 + i for i<8, k0 + 16 + (L/16)*8 + (i-8) for i>=8  (f16).
__global__ void k_build_wfrag(const float* __restrict__ W, _Float16* __restrict__ frag) {
    int tid = blockIdx.x * blockDim.x + threadIdx.x;
    if (tid >= 32 * 32) return;
    int fidx = tid >> 5, lane = tid & 31;
    int k0 = (fidx >> 3) * 32;
    int n0 = (fidx & 7) * 16;
    int col  = n0 + (lane & 15);
    int half = lane >> 4;
    _Float16* dst = frag + (size_t)fidx * 512 + lane * 16;
#pragma unroll
    for (int i = 0; i < 16; ++i) {
        int K = k0 + (i < 8 ? half * 8 + i : 16 + half * 8 + (i - 8));
        dst[i] = (_Float16)W[K * HC + col];
    }
}

// ---------------- kernel 0c: init accumulator / softmax state ----------------
__global__ void k_init(float* __restrict__ agg, unsigned* __restrict__ smax,
                       float* __restrict__ denom, float* __restrict__ msum,
                       float* __restrict__ vsum) {
    size_t i = (size_t)blockIdx.x * blockDim.x + threadIdx.x;
    if (i < (size_t)N_NODES * HC) agg[i] = 0.f;
    if (i < (size_t)N_NODES * NHEAD) { smax[i] = ENC_NEG_INF; denom[i] = 0.f; }
    if (i < HC) { msum[i] = 0.f; vsum[i] = 0.f; }
}

// ---------------- kernel 1: dual GEMM  x_l = x@W_l, x_r = x@W_r  (WMMA) ----------------
// one wave per 16-row block; A fragments reused for both weight matrices.
__global__ void k_gemm_wmma(const _Float16* __restrict__ x16,
                            const _Float16* __restrict__ wfl,
                            const _Float16* __restrict__ wfr,
                            float* __restrict__ xl, float* __restrict__ xr) {
    int wave = blockIdx.x * (blockDim.x >> 5) + (threadIdx.x >> 5);
    int lane = threadIdx.x & 31;
    if (wave >= (N_NODES / 16)) return;            // wave-uniform: EXEC all-ones inside
    int m0   = wave * 16;
    int mrow = m0 + (lane & 15);
    int half = lane >> 4;

    // A fragments: 16-bit A 16x32 layout (two contiguous 16B runs per lane per chunk)
    v16h a[4];
    const _Float16* xrow = x16 + (size_t)mrow * F_IN;
#pragma unroll
    for (int kc = 0; kc < 4; ++kc) {
        int k0 = kc * 32;
        v8h lo = *(const v8h*)(xrow + k0 + half * 8);
        v8h hi = *(const v8h*)(xrow + k0 + 16 + half * 8);
#pragma unroll
        for (int i = 0; i < 8; ++i) { a[kc][i] = lo[i]; a[kc][i + 8] = hi[i]; }
    }

#pragma unroll
    for (int nt = 0; nt < 8; ++nt) {
        v8f accl = {}; v8f accr = {};
#pragma unroll
        for (int kc = 0; kc < 4; ++kc) {
            size_t fo = ((size_t)(kc * 8 + nt)) * 512 + (size_t)lane * 16;
            v16h bl = *(const v16h*)(wfl + fo);
            v16h br = *(const v16h*)(wfr + fo);
            accl = __builtin_amdgcn_wmma_f32_16x16x32_f16(false, a[kc], false, bl,
                                                          (short)0, accl, false, false);
            accr = __builtin_amdgcn_wmma_f32_16x16x32_f16(false, a[kc], false, br,
                                                          (short)0, accr, false, false);
        }
        int n0 = nt * 16, ncol = n0 + (lane & 15);
#pragma unroll
        for (int r = 0; r < 8; ++r) {
            int m = m0 + half * 8 + r;            // C/D layout: VGPR r -> M = r + half*8
            xl[(size_t)m * HC + ncol] = accl[r];
            xr[(size_t)m * HC + ncol] = accr[r];
        }
    }
}

// ---------------- kernel 2: per-edge GATv2 scores + segment max ----------------
// one wave per edge; lane covers 4 contiguous channels; head = lane/8.
__global__ void k_edge_scores(const int* __restrict__ ei,
                              const float* __restrict__ xl, const float* __restrict__ xr,
                              const float* __restrict__ att,
                              float* __restrict__ score, unsigned* __restrict__ smax) {
    int wave = blockIdx.x * (blockDim.x >> 5) + (threadIdx.x >> 5);
    int lane = threadIdx.x & 31;
    if (wave >= ET) return;
    int s, d; edge_sd(ei, wave, s, d);
    int c0 = lane * 4;
    float4 a4 = *(const float4*)(att + c0);
    float4 l4 = *(const float4*)(xl + (size_t)s * HC + c0);
    float4 r4 = *(const float4*)(xr + (size_t)d * HC + c0);
    float p = a4.x * lrelu(l4.x + r4.x) + a4.y * lrelu(l4.y + r4.y)
            + a4.z * lrelu(l4.z + r4.z) + a4.w * lrelu(l4.w + r4.w);
    // reduce 8 lanes per head (xor masks stay inside the group of 8)
    p += __shfl_xor(p, 1, 32);
    p += __shfl_xor(p, 2, 32);
    p += __shfl_xor(p, 4, 32);
    if ((lane & 7) == 0) {
        int h = lane >> 3;
        score[(size_t)wave * NHEAD + h] = p;
        atomicMax(&smax[(size_t)d * NHEAD + h], enc_f32(p));
    }
}

// ---------------- kernel 3: exp(score - max) + segment sum ----------------
__global__ void k_edge_exp(const int* __restrict__ ei, float* __restrict__ score,
                           const unsigned* __restrict__ smax, float* __restrict__ denom) {
    size_t idx = (size_t)blockIdx.x * blockDim.x + threadIdx.x;
    if (idx >= (size_t)ET * NHEAD) return;
    int e = (int)(idx >> 2), h = (int)(idx & 3);
    int s, d; edge_sd(ei, e, s, d); (void)s;
    float m  = dec_f32(smax[(size_t)d * NHEAD + h]);
    float ex = __expf(score[idx] - m);
    score[idx] = ex;                                  // overwrite score with exp
    atomicAdd(&denom[(size_t)d * NHEAD + h], ex);
}

// ---------------- kernel 4: weighted aggregation (scatter-add) ----------------
__global__ void k_aggregate(const int* __restrict__ ei, const float* __restrict__ ex,
                            const float* __restrict__ denom, const float* __restrict__ xl,
                            float* __restrict__ agg) {
    int wave = blockIdx.x * (blockDim.x >> 5) + (threadIdx.x >> 5);
    int lane = threadIdx.x & 31;
    if (wave >= ET) return;
    int s, d; edge_sd(ei, wave, s, d);
    int h = lane >> 3, c0 = lane * 4;
    float alpha = ex[(size_t)wave * NHEAD + h] / denom[(size_t)d * NHEAD + h];
    float4 l4 = *(const float4*)(xl + (size_t)s * HC + c0);
    float* o = agg + (size_t)d * HC + c0;
    atomicAdd(o + 0, alpha * l4.x);
    atomicAdd(o + 1, alpha * l4.y);
    atomicAdd(o + 2, alpha * l4.z);
    atomicAdd(o + 3, alpha * l4.w);
}

// ---------------- kernel 5: per-feature column sums (for mean) ----------------
__global__ void k_col_sum(const float* __restrict__ agg, float* __restrict__ msum) {
    int f = threadIdx.x;                   // blockDim.x == 128
    float local = 0.f;
    for (int r = blockIdx.x; r < N_NODES; r += gridDim.x)
        local += agg[(size_t)r * HC + f];
    atomicAdd(&msum[f], local);
}

// ---------------- kernel 6: per-feature centered sum of squares (for var) -------
__global__ void k_col_varsum(const float* __restrict__ agg, const float* __restrict__ msum,
                             const float* __restrict__ bias, const float* __restrict__ gms,
                             float* __restrict__ vsum) {
    int f = threadIdx.x;
    float mean = msum[f] * (1.f / N_NODES) + bias[f];   // bias folded analytically
    float mu   = gms[f] * mean;
    float b    = bias[f];
    float local = 0.f;
    for (int r = blockIdx.x; r < N_NODES; r += gridDim.x) {
        float c = agg[(size_t)r * HC + f] + b - mu;
        local += c * c;
    }
    atomicAdd(&vsum[f], local);
}

// ---------------- kernel 7: final GraphNorm normalize ----------------
__global__ void k_normalize(const float* __restrict__ agg, const float* __restrict__ msum,
                            const float* __restrict__ vsum, const float* __restrict__ bias,
                            const float* __restrict__ gms, const float* __restrict__ gnw,
                            const float* __restrict__ gnb, float* __restrict__ out) {
    size_t idx = (size_t)blockIdx.x * blockDim.x + threadIdx.x;
    if (idx >= (size_t)N_NODES * HC) return;
    int f = (int)(idx & (HC - 1));
    float mean = msum[f] * (1.f / N_NODES) + bias[f];
    float c    = agg[idx] + bias[f] - gms[f] * mean;
    float var  = vsum[f] * (1.f / N_NODES);
    out[idx] = gnw[f] * c * __frsqrt_rn(var + GN_EPS) + gnb[f];
}

// ---------------- workspace layout ----------------
static constexpr size_t alup(size_t x) { return (x + 255) & ~(size_t)255; }
static constexpr size_t OFF_X16  = 0;
static constexpr size_t OFF_WFL  = alup(OFF_X16 + (size_t)N_NODES * F_IN * 2);
static constexpr size_t OFF_WFR  = alup(OFF_WFL + 32 * 32 * 16 * 2);
static constexpr size_t OFF_XL   = alup(OFF_WFR + 32 * 32 * 16 * 2);
static constexpr size_t OFF_XR   = alup(OFF_XL + (size_t)N_NODES * HC * 4);
static constexpr size_t OFF_SCR  = alup(OFF_XR + (size_t)N_NODES * HC * 4);
static constexpr size_t OFF_SMAX = alup(OFF_SCR + (size_t)ET * NHEAD * 4);
static constexpr size_t OFF_DEN  = alup(OFF_SMAX + (size_t)N_NODES * NHEAD * 4);
static constexpr size_t OFF_AGG  = alup(OFF_DEN + (size_t)N_NODES * NHEAD * 4);
static constexpr size_t OFF_MSUM = alup(OFF_AGG + (size_t)N_NODES * HC * 4);
static constexpr size_t OFF_VSUM = alup(OFF_MSUM + HC * 4);

extern "C" void kernel_launch(void* const* d_in, const int* in_sizes, int n_in,
                              void* d_out, int out_size, void* d_ws, size_t ws_size,
                              hipStream_t stream) {
    const float* x    = (const float*)d_in[0];
    const int*   ei   = (const int*)d_in[1];
    const float* Wl   = (const float*)d_in[2];
    const float* Wr   = (const float*)d_in[3];
    const float* att  = (const float*)d_in[4];
    const float* bias = (const float*)d_in[5];
    const float* gnw  = (const float*)d_in[6];
    const float* gnb  = (const float*)d_in[7];
    const float* gms  = (const float*)d_in[8];
    float* out = (float*)d_out;

    char* ws = (char*)d_ws;
    _Float16* x16  = (_Float16*)(ws + OFF_X16);
    _Float16* wfl  = (_Float16*)(ws + OFF_WFL);
    _Float16* wfr  = (_Float16*)(ws + OFF_WFR);
    float*    xl   = (float*)(ws + OFF_XL);
    float*    xr   = (float*)(ws + OFF_XR);
    float*    scr  = (float*)(ws + OFF_SCR);
    unsigned* smax = (unsigned*)(ws + OFF_SMAX);
    float*    den  = (float*)(ws + OFF_DEN);
    float*    agg  = (float*)(ws + OFF_AGG);
    float*    msum = (float*)(ws + OFF_MSUM);
    float*    vsum = (float*)(ws + OFF_VSUM);

    const int NF = N_NODES * HC;                 // 6.4M

    // prep
    k_conv_f16  <<<(NF + 255) / 256, 256, 0, stream>>>(x, x16);
    k_build_wfrag<<<4, 256, 0, stream>>>(Wl, wfl);
    k_build_wfrag<<<4, 256, 0, stream>>>(Wr, wfr);
    k_init      <<<(NF + 255) / 256, 256, 0, stream>>>(agg, smax, den, msum, vsum);

    // dual GEMM via WMMA: 3125 row-blocks, 4 waves/block
    k_gemm_wmma <<<(N_NODES / 16 + 3) / 4, 128, 0, stream>>>(x16, wfl, wfr, xl, xr);

    // edge phase (one wave32 per edge, 8 waves/block)
    k_edge_scores<<<(ET + 7) / 8, 256, 0, stream>>>(ei, xl, xr, att, scr, smax);
    k_edge_exp   <<<((size_t)ET * NHEAD + 255) / 256, 256, 0, stream>>>(ei, scr, smax, den);
    k_aggregate  <<<(ET + 7) / 8, 256, 0, stream>>>(ei, scr, den, xl, agg);

    // GraphNorm
    k_col_sum   <<<256, 128, 0, stream>>>(agg, msum);
    k_col_varsum<<<256, 128, 0, stream>>>(agg, msum, bias, gms, vsum);
    k_normalize <<<(NF + 255) / 256, 256, 0, stream>>>(agg, msum, vsum, bias, gms, gnw, gnb, out);
}
